// RNNBlock_43559558316460
// MI455X (gfx1250) — matrix operand order repaired and checked by reading, not compiled, not run on hardware
//
#include <hip/hip_runtime.h>
#include <hip/hip_bf16.h>
#include <stdint.h>

// ---------------------------------------------------------------------------
// Bidirectional 2-layer LSTM (H=512, Cin=256, B=128, T=256) + 1024->80 proj.
// bf16 WMMA (v_wmma_f32_16x16x32_bf16) everywhere; f32 accumulation.
// Persistent batch-sliced recurrence: 16 WGs (8 batch slices x 2 directions),
// zero cross-WG sync; weights stream from L2 (all fit in 192MB), activations
// live in LDS, cell state lives in registers for all 256 steps.
// ---------------------------------------------------------------------------

#define HDIM   512
#define CIN    256
#define TSEQ   256
#define BATCH  128
#define NCLS   80
#define G4H    2048              // 4*H gate width
#define LDSTR  1288              // LDS row stride in bf16 elems (16B aligned, bank-spread)

typedef __attribute__((ext_vector_type(16))) __bf16 v16bf;
typedef __attribute__((ext_vector_type(8)))  float  v8f;

union FragBF {
    v16bf v;
    uint4 u[2];
};

__device__ __forceinline__ v8f vzero8() {
    v8f z;
#pragma unroll
    for (int i = 0; i < 8; ++i) z[i] = 0.0f;
    return z;
}

// Fast gate nonlinearities built on the native v_exp_f32 (TRANS pipe,
// co-executes with the matrix pipe).  sig: exp2(-x*log2e) saturates cleanly
// to 0/1.  tanh: clamp first so e stays finite (no inf/inf NaN).
__device__ __forceinline__ float fast_sig(float x) {
    return 1.0f / (1.0f + __builtin_amdgcn_exp2f(-1.44269504f * x));
}
__device__ __forceinline__ float fast_tanh(float x) {
    x = fminf(fmaxf(x, -20.0f), 20.0f);
    float e = __builtin_amdgcn_exp2f(2.88539008f * x);   // exp(2x)
    return (e - 1.0f) / (e + 1.0f);
}

// --------------------------- elementwise prep kernels ----------------------

__global__ void convert_x_kernel(const float* __restrict__ x,
                                 __bf16* __restrict__ xb, int n) {
    int i = blockIdx.x * blockDim.x + threadIdx.x;
    if (i < n) xb[i] = (__bf16)x[i];
}

// dst[n'][Ktot] bf16, n' gate-quartet permuted:
//   quad = n'>>4, gate = quad&3, hTile = quad>>2, within = n'&15
//   src column c = gate*512 + hTile*16 + within
// rows k<Kx come from Wx[k][c], rows k>=Kx from Wh[k-Kx][c]   (both [K][2048])
__global__ void build_w_kernel(const float* __restrict__ Wx,
                               const float* __restrict__ Wh,
                               __bf16* __restrict__ dst, int Kx, int Ktot) {
    long long idx = (long long)blockIdx.x * blockDim.x + threadIdx.x;
    long long total = (long long)G4H * Ktot;
    if (idx >= total) return;
    int k  = (int)(idx % Ktot);
    int np = (int)(idx / Ktot);
    int within = np & 15;
    int quad   = np >> 4;
    int gate   = quad & 3;
    int hT     = quad >> 2;
    int c = gate * HDIM + hT * 16 + within;
    float v = (k < Kx) ? Wx[(long long)k * G4H + c]
                       : Wh[(long long)(k - Kx) * G4H + c];
    dst[idx] = (__bf16)v;
}

// dst[80][1024] bf16 <- transpose of Wproj [1024][80] f32
__global__ void build_wproj_kernel(const float* __restrict__ Wp,
                                   __bf16* __restrict__ dst) {
    int idx = blockIdx.x * blockDim.x + threadIdx.x;
    if (idx >= NCLS * 2 * HDIM) return;
    int k = idx & (2 * HDIM - 1);
    int n = idx >> 10;
    dst[idx] = (__bf16)Wp[(long long)k * NCLS + n];
}

// --------------------------- persistent recurrent kernel -------------------
// grid = 16 blocks (8 batch slices x 2 directions), block = 512 threads (16 waves).
// Each wave owns hidden tiles {2w, 2w+1}  ->  z tiles 8w..8w+7 (i,f,g,o quartets).
// LDS activation buffer per slice: cols [0,256)=x_t  [256,768)=h1  [768,1280)=h2.

__global__ __launch_bounds__(512) void lstm_seq_kernel(
    const __bf16* __restrict__ xbf,                    // [B][T][CIN]
    const __bf16* __restrict__ W1f, const __bf16* __restrict__ W2f,
    const __bf16* __restrict__ W1b, const __bf16* __restrict__ W2b,
    __bf16* __restrict__ seq)                          // [T*B][2H]: fwd|bwd
{
    __shared__ __bf16 act[16 * LDSTR];

    const int tid  = threadIdx.x;
    const int wave = tid >> 5;
    const int lane = tid & 31;
    const int col  = lane & 15;
    const int half = lane >> 4;
    const int dir  = blockIdx.x >> 3;
    const int b0   = (blockIdx.x & 7) * 16;

    const __bf16* __restrict__ W1 = dir ? W1b : W1f;
    const __bf16* __restrict__ W2 = dir ? W2b : W2f;

    // zero h1/h2 regions
    for (int i = tid; i < 16 * 1024; i += 512) {
        int r = i >> 10, c = i & 1023;
        act[r * LDSTR + 256 + c] = (__bf16)0.0f;
    }

    float c1s[2][8], c2s[2][8];
#pragma unroll
    for (int j = 0; j < 2; ++j)
#pragma unroll
        for (int v = 0; v < 8; ++v) { c1s[j][v] = 0.0f; c2s[j][v] = 0.0f; }

    for (int s = 0; s < TSEQ; ++s) {
        const int t = dir ? (TSEQ - 1 - s) : s;

        // stage x_t: 16 rows x 256 cols bf16, one uint4 (8 elems) per thread.
        // (No barrier needed before this: previous iteration's readers of the
        //  x region are >=3 barriers behind.)
        {
            int r  = tid >> 5;
            int ch = (tid & 31) * 8;
            *(uint4*)(&act[r * LDSTR + ch]) =
                *(const uint4*)(xbf + ((long long)(b0 + r) * TSEQ + t) * CIN + ch);
        }
        __syncthreads();

        // ----- layer 1: z1 = [x_t, h1] @ W1   (K = 768) -----
        v8f acc[8];
#pragma unroll
        for (int tt = 0; tt < 8; ++tt) acc[tt] = vzero8();

        for (int k = 0; k < CIN + HDIM; k += 32) {
            FragBF fa;
            {
                const __bf16* p = &act[col * LDSTR + k + half * 8];
                fa.u[0] = *(const uint4*)(p);
                fa.u[1] = *(const uint4*)(p + 16);
            }
#pragma unroll
            for (int tt = 0; tt < 8; ++tt) {
                FragBF fb;
                const __bf16* p =
                    W1 + (long long)((8 * wave + tt) * 16 + col) * (CIN + HDIM)
                       + k + half * 8;
                fb.u[0] = *(const uint4*)(p);
                fb.u[1] = *(const uint4*)(p + 16);
                acc[tt] = __builtin_amdgcn_wmma_f32_16x16x32_bf16(
                    false, fa.v, false, fb.v, (short)0, acc[tt], false, false);
            }
        }
        __syncthreads();   // everyone done reading old h1

        // gates layer 1 -> h1 (LDS)
#pragma unroll
        for (int jj = 0; jj < 2; ++jj) {
            int hTile = 2 * wave + jj;
#pragma unroll
            for (int v = 0; v < 8; ++v) {
                float zi = acc[4 * jj + 0][v];
                float zf = acc[4 * jj + 1][v];
                float zg = acc[4 * jj + 2][v];
                float zo = acc[4 * jj + 3][v];
                float c = fast_sig(zf) * c1s[jj][v] + fast_sig(zi) * fast_tanh(zg);
                c1s[jj][v] = c;
                float h = fast_sig(zo) * fast_tanh(c);
                int M = v + 8 * half;
                act[M * LDSTR + 256 + hTile * 16 + col] = (__bf16)h;
            }
        }
        __syncthreads();

        // ----- layer 2: z2 = [h1, h2] @ W2   (K = 1024, A offset 256) -----
#pragma unroll
        for (int tt = 0; tt < 8; ++tt) acc[tt] = vzero8();

        for (int k = 0; k < 2 * HDIM; k += 32) {
            FragBF fa;
            {
                const __bf16* p = &act[col * LDSTR + 256 + k + half * 8];
                fa.u[0] = *(const uint4*)(p);
                fa.u[1] = *(const uint4*)(p + 16);
            }
#pragma unroll
            for (int tt = 0; tt < 8; ++tt) {
                FragBF fb;
                const __bf16* p =
                    W2 + (long long)((8 * wave + tt) * 16 + col) * (2 * HDIM)
                       + k + half * 8;
                fb.u[0] = *(const uint4*)(p);
                fb.u[1] = *(const uint4*)(p + 16);
                acc[tt] = __builtin_amdgcn_wmma_f32_16x16x32_bf16(
                    false, fa.v, false, fb.v, (short)0, acc[tt], false, false);
            }
        }
        __syncthreads();   // everyone done reading old h2

        // gates layer 2 -> h2 (LDS) + sequence output (global, bf16)
#pragma unroll
        for (int jj = 0; jj < 2; ++jj) {
            int hTile = 2 * wave + jj;
#pragma unroll
            for (int v = 0; v < 8; ++v) {
                float zi = acc[4 * jj + 0][v];
                float zf = acc[4 * jj + 1][v];
                float zg = acc[4 * jj + 2][v];
                float zo = acc[4 * jj + 3][v];
                float c = fast_sig(zf) * c2s[jj][v] + fast_sig(zi) * fast_tanh(zg);
                c2s[jj][v] = c;
                float h = fast_sig(zo) * fast_tanh(c);
                int M = v + 8 * half;
                __bf16 hb = (__bf16)h;
                act[M * LDSTR + 768 + hTile * 16 + col] = hb;
                seq[((long long)t * BATCH + b0 + M) * (2 * HDIM)
                    + dir * HDIM + hTile * 16 + col] = hb;
            }
        }
        __syncthreads();   // h2 visible before next step's layer-2 reads
    }
}

// --------------------------- projection kernel -----------------------------
// out[T*B][80] = seq[T*B][1024] @ Wproj[1024][80].  One M-tile per wave,
// 5 N-tiles (=80 cols) per wave, K = 1024 in 32 WMMA steps.

__global__ __launch_bounds__(256) void proj_kernel(
    const __bf16* __restrict__ seq,   // [32768][1024]
    const __bf16* __restrict__ Wpt,   // [80][1024]  (transposed)
    float* __restrict__ out)          // [32768][80]
{
    const int wave = threadIdx.x >> 5;
    const int lane = threadIdx.x & 31;
    const int col  = lane & 15;
    const int half = lane >> 4;
    const int mTile = blockIdx.x * 8 + wave;      // 0..2047

    v8f acc[5];
#pragma unroll
    for (int nt = 0; nt < 5; ++nt) acc[nt] = vzero8();

    for (int k = 0; k < 2 * HDIM; k += 32) {
        FragBF fa;
        {
            const __bf16* p =
                seq + ((long long)mTile * 16 + col) * (2 * HDIM) + k + half * 8;
            fa.u[0] = *(const uint4*)(p);
            fa.u[1] = *(const uint4*)(p + 16);
        }
#pragma unroll
        for (int nt = 0; nt < 5; ++nt) {
            FragBF fb;
            const __bf16* p =
                Wpt + (long long)(nt * 16 + col) * (2 * HDIM) + k + half * 8;
            fb.u[0] = *(const uint4*)(p);
            fb.u[1] = *(const uint4*)(p + 16);
            acc[nt] = __builtin_amdgcn_wmma_f32_16x16x32_bf16(
                false, fa.v, false, fb.v, (short)0, acc[nt], false, false);
        }
    }

#pragma unroll
    for (int nt = 0; nt < 5; ++nt)
#pragma unroll
        for (int v = 0; v < 8; ++v) {
            int M = v + 8 * half;
            out[((long long)mTile * 16 + M) * NCLS + nt * 16 + col] = acc[nt][v];
        }
}

// --------------------------- launch ----------------------------------------

extern "C" void kernel_launch(void* const* d_in, const int* in_sizes, int n_in,
                              void* d_out, int out_size, void* d_ws, size_t ws_size,
                              hipStream_t stream) {
    const float* x     = (const float*)d_in[0];
    const float* Wx_f1 = (const float*)d_in[1];
    const float* Wh_f1 = (const float*)d_in[2];
    const float* Wx_f2 = (const float*)d_in[3];
    const float* Wh_f2 = (const float*)d_in[4];
    const float* Wx_b1 = (const float*)d_in[5];
    const float* Wh_b1 = (const float*)d_in[6];
    const float* Wx_b2 = (const float*)d_in[7];
    const float* Wh_b2 = (const float*)d_in[8];
    const float* Wproj = (const float*)d_in[9];
    float* out = (float*)d_out;

    // workspace layout (bf16 elements), total ~99 MB
    constexpr long long N_XBF = (long long)BATCH * TSEQ * CIN;        // 8,388,608
    constexpr long long N_W1  = (long long)G4H * (CIN + HDIM);        // 1,572,864
    constexpr long long N_W2  = (long long)G4H * (2 * HDIM);          // 2,097,152
    constexpr long long N_WP  = (long long)NCLS * (2 * HDIM);         //    81,920

    __bf16* ws  = (__bf16*)d_ws;
    __bf16* xbf = ws;
    __bf16* w1f = xbf + N_XBF;
    __bf16* w2f = w1f + N_W1;
    __bf16* w1b = w2f + N_W2;
    __bf16* w2b = w1b + N_W1;
    __bf16* wpt = w2b + N_W2;
    __bf16* seq = wpt + N_WP;

    // 1) convert x to bf16
    {
        int n = (int)N_XBF;
        convert_x_kernel<<<(n + 255) / 256, 256, 0, stream>>>(x, xbf, n);
    }
    // 2) build transposed + gate-permuted bf16 weights
    {
        int b1 = (int)((N_W1 + 255) / 256);
        int b2 = (int)((N_W2 + 255) / 256);
        build_w_kernel<<<b1, 256, 0, stream>>>(Wx_f1, Wh_f1, w1f, CIN, CIN + HDIM);
        build_w_kernel<<<b2, 256, 0, stream>>>(Wx_f2, Wh_f2, w2f, HDIM, 2 * HDIM);
        build_w_kernel<<<b1, 256, 0, stream>>>(Wx_b1, Wh_b1, w1b, CIN, CIN + HDIM);
        build_w_kernel<<<b2, 256, 0, stream>>>(Wx_b2, Wh_b2, w2b, HDIM, 2 * HDIM);
        build_wproj_kernel<<<(int)((N_WP + 255) / 256), 256, 0, stream>>>(Wproj, wpt);
    }
    // 3) persistent recurrent kernel: 8 batch slices x 2 directions
    lstm_seq_kernel<<<16, 512, 0, stream>>>(xbf, w1f, w2f, w1b, w2b, seq);

    // 4) projection: 2048 M-tiles, 8 waves per block
    proj_kernel<<<256, 256, 0, stream>>>(seq, wpt, out);
}
